// GNNDenoiser_30425548325379
// MI455X (gfx1250) — matrix operand order repaired and checked by reading, not compiled, not run on hardware
//
#include <hip/hip_runtime.h>
#include <hip/hip_bf16.h>

// GNN denoiser for MI455X (gfx1250, wave32). All GEMMs run on
// v_wmma_f32_16x16x32_bf16 (bf16 A/B, f32 accum). Workload is compute-limited
// (~225 GFLOP vs ~0.5 GB HBM traffic). This revision adds M-blocking (2 row
// tiles per wave share each B tile) to double WMMA-per-weight-load.

typedef __attribute__((ext_vector_type(16))) __bf16 v16bf;
typedef __attribute__((ext_vector_type(8)))  float  v8f;

#define HID 128
#define NLAYER 4
#define CODE 1024

static __device__ inline v8f wmma_bf16(v16bf a, v16bf b, v8f c) {
  // 8 args: (neg_a, A, neg_b, B, c_mod, C, reuse_a, reuse_b)
  return __builtin_amdgcn_wmma_f32_16x16x32_bf16(false, a, false, b,
                                                 (short)0, c, false, false);
}

static __device__ inline v8f zero8() {
  v8f z;
#pragma unroll
  for (int i = 0; i < 8; ++i) z[i] = 0.f;
  return z;
}

static __device__ inline float silu(float x) { return x / (1.f + __expf(-x)); }

// 16-bit A-tile (16x32, MxK) layout, wave32:
//   lane = (m = lane&15, half = lane>>4); element j<8  -> K = half*8 + j
//                                         element j>=8 -> K = 16 + half*8 + (j-8)
static __device__ inline v16bf load_a16(const float* rowk, int half) {
  v16bf a;
  const float* p0 = rowk + half * 8;
  const float* p1 = rowk + 16 + half * 8;
#pragma unroll
  for (int j = 0; j < 8; ++j) a[j] = (__bf16)p0[j];
#pragma unroll
  for (int j = 0; j < 8; ++j) a[8 + j] = (__bf16)p1[j];
  return a;
}

static __device__ inline v16bf load_a16s(const float* rowk, int half, float s) {
  v16bf a;
  const float* p0 = rowk + half * 8;
  const float* p1 = rowk + 16 + half * 8;
#pragma unroll
  for (int j = 0; j < 8; ++j) a[j] = (__bf16)(p0[j] * s);
#pragma unroll
  for (int j = 0; j < 8; ++j) a[8 + j] = (__bf16)(p1[j] * s);
  return a;
}

// 16-bit B-tile (32x16, KxN): lane = (n = lane&15, half); element j<8 -> K = half*8 + j,
// element j>=8 -> K = 16 + half*8 + (j-8).
// B[k][n] = W[n][k] for out = in @ W^T, W row-major [out][in] (bf16).
static __device__ inline v16bf load_b16(const __bf16* W, int ldk, int n, int kb, int half) {
  const __bf16* p = W + (size_t)n * ldk + kb + half * 8;
  v16bf b;
#pragma unroll
  for (int j = 0; j < 8; ++j) b[j] = p[j];
#pragma unroll
  for (int j = 0; j < 8; ++j) b[8 + j] = p[16 + j];
  return b;
}

// ---------------------------------------------------------------- weight cvt
__global__ void k_convert(const float* __restrict__ src, __bf16* __restrict__ dst,
                          int rows, int src_ld, int cols) {
  int i = blockIdx.x * blockDim.x + threadIdx.x;
  if (i < rows * cols) {
    int r = i / cols, c = i - r * cols;
    dst[i] = (__bf16)src[(size_t)r * src_ld + c];
  }
}

// ------------------------------------------------------- edge geometry + cnt
__global__ void k_edge_geom(const int* __restrict__ ei, const float* __restrict__ coords,
                            float* __restrict__ dist, float* __restrict__ cnt, int E) {
  int e = blockIdx.x * blockDim.x + threadIdx.x;
  if (e >= E) return;
  int r = ei[e], c = ei[(size_t)E + e];
  float dx = coords[c * 3 + 0] - coords[r * 3 + 0];
  float dy = coords[c * 3 + 1] - coords[r * 3 + 1];
  float dz = coords[c * 3 + 2] - coords[r * 3 + 2];
  dist[e] = sqrtf(dx * dx + dy * dy + dz * dz);
  atomicAdd(&cnt[c], 1.f);
}

// ------------------------------------------------------------- input GEMM
// h[N,128] = y[N,1024] @ W_in^T + b_in. Wave: 32 rows x 128 cols, K=1024.
__global__ __launch_bounds__(256) void k_gemm_in(const float* __restrict__ y,
                                                 const __bf16* __restrict__ W,
                                                 const float* __restrict__ bias,
                                                 float* __restrict__ h, int N) {
  const int lane = threadIdx.x & 31, wv = threadIdx.x >> 5;
  const int m = lane & 15, half = lane >> 4;
  const int wave = blockIdx.x * (blockDim.x >> 5) + wv;
  const int waves = N >> 5;
  if (wave >= waves) return;
  const int base = wave * 32;
  const float* rowA0 = y + (size_t)(base + m) * CODE;
  const float* rowA1 = y + (size_t)(base + 16 + m) * CODE;
  v8f acc[2][8];
#pragma unroll
  for (int mb = 0; mb < 2; ++mb)
#pragma unroll
    for (int nt = 0; nt < 8; ++nt) acc[mb][nt] = zero8();
#pragma unroll 2
  for (int kb = 0; kb < CODE; kb += 32) {
    v16bf a0 = load_a16(rowA0 + kb, half);
    v16bf a1 = load_a16(rowA1 + kb, half);
#pragma unroll
    for (int nt = 0; nt < 8; ++nt) {
      v16bf b = load_b16(W, CODE, nt * 16 + m, kb, half);
      acc[0][nt] = wmma_bf16(a0, b, acc[0][nt]);
      acc[1][nt] = wmma_bf16(a1, b, acc[1][nt]);
    }
  }
#pragma unroll
  for (int mb = 0; mb < 2; ++mb)
#pragma unroll
    for (int nt = 0; nt < 8; ++nt) {
      const int n = nt * 16 + m;
      const float bi = bias[n];
#pragma unroll
      for (int r = 0; r < 8; ++r) {
        int nd = base + mb * 16 + r + 8 * half;  // C layout: M = r + 8*half
        h[(size_t)nd * HID + n] = acc[mb][nt][r] + bi;
      }
    }
}

// ------------------------------------------------------------- layernorm
__global__ __launch_bounds__(256) void k_layernorm(const float* __restrict__ h,
                                                   float* __restrict__ hn,
                                                   const float* __restrict__ gam,
                                                   const float* __restrict__ bet, int N) {
  const int lane = threadIdx.x & 31, wv = threadIdx.x >> 5;
  const int node = blockIdx.x * (blockDim.x >> 5) + wv;
  if (node >= N) return;
  const float* row = h + (size_t)node * HID;
  float v[4], s = 0.f, s2 = 0.f;
#pragma unroll
  for (int j = 0; j < 4; ++j) {
    v[j] = row[lane + 32 * j];
    s += v[j];
    s2 += v[j] * v[j];
  }
#pragma unroll
  for (int msk = 16; msk >= 1; msk >>= 1) {
    s += __shfl_xor(s, msk, 32);
    s2 += __shfl_xor(s2, msk, 32);
  }
  const float mu = s * (1.f / HID);
  const float var = s2 * (1.f / HID) - mu * mu;
  const float rs = rsqrtf(var + 1e-5f);
#pragma unroll
  for (int j = 0; j < 4; ++j) {
    int k = lane + 32 * j;
    hn[(size_t)node * HID + k] = (v[j] - mu) * rs * gam[k] + bet[k];
  }
}

// ------------------------------------------------------------- edge MLP (fused)
// per 32-edge wave tile: m = silu(silu([hn[row],hn[col],dist]@W1^T+b1)@W2^T+b2);
// scatter-mean accumulate into m_aggr via f32 atomics. 2 waves/block, 32KB LDS.
__global__ __launch_bounds__(64) void k_edge(const float* __restrict__ hn,
                                             const int* __restrict__ ei,
                                             const float* __restrict__ dist,
                                             const __bf16* __restrict__ W1,
                                             const float* __restrict__ W1f,
                                             const float* __restrict__ b1,
                                             const __bf16* __restrict__ W2,
                                             const float* __restrict__ b2,
                                             float* __restrict__ maggr, int E) {
  __shared__ float smem[2 * 32 * HID];  // per-wave 32x128 fp32 re-layout buffer
  const int lane = threadIdx.x & 31, wv = threadIdx.x >> 5;
  const int m = lane & 15, half = lane >> 4;
  const int e_base = (blockIdx.x * 2 + wv) * 32;

  int em0 = e_base + m;      if (em0 >= E) em0 = E - 1;
  int em1 = e_base + 16 + m; if (em1 >= E) em1 = E - 1;
  const float* rowR0 = hn + (size_t)ei[em0] * HID;
  const float* rowR1 = hn + (size_t)ei[em1] * HID;
  const float* rowC0 = hn + (size_t)ei[(size_t)E + em0] * HID;
  const float* rowC1 = hn + (size_t)ei[(size_t)E + em1] * HID;

  v8f acc[2][8];
#pragma unroll
  for (int mb = 0; mb < 2; ++mb)
#pragma unroll
    for (int nt = 0; nt < 8; ++nt) acc[mb][nt] = zero8();

#pragma unroll
  for (int kb = 0; kb < HID; kb += 32) {
    v16bf a0 = load_a16(rowR0 + kb, half);
    v16bf a1 = load_a16(rowR1 + kb, half);
#pragma unroll
    for (int nt = 0; nt < 8; ++nt) {
      v16bf b = load_b16(W1, 256, nt * 16 + m, kb, half);
      acc[0][nt] = wmma_bf16(a0, b, acc[0][nt]);
      acc[1][nt] = wmma_bf16(a1, b, acc[1][nt]);
    }
  }
#pragma unroll
  for (int kb = 0; kb < HID; kb += 32) {
    v16bf a0 = load_a16(rowC0 + kb, half);
    v16bf a1 = load_a16(rowC1 + kb, half);
#pragma unroll
    for (int nt = 0; nt < 8; ++nt) {
      v16bf b = load_b16(W1, 256, nt * 16 + m, HID + kb, half);
      acc[0][nt] = wmma_bf16(a0, b, acc[0][nt]);
      acc[1][nt] = wmma_bf16(a1, b, acc[1][nt]);
    }
  }

  float* my = smem + wv * 32 * HID;
#pragma unroll
  for (int mb = 0; mb < 2; ++mb) {
    // C-tile row r of block mb holds edge (e_base + mb*16 + r + 8*half)
    float dr[8];
#pragma unroll
    for (int r = 0; r < 8; ++r) {
      int e = e_base + mb * 16 + r + 8 * half;
      dr[r] = (e < E) ? dist[e] : 0.f;
    }
#pragma unroll
    for (int nt = 0; nt < 8; ++nt) {
      const int n = nt * 16 + m;
      const float wl = W1f[(size_t)n * 257 + 256];  // dist column (kept fp32)
      const float bb = b1[n];
#pragma unroll
      for (int r = 0; r < 8; ++r)
        my[(mb * 16 + r + 8 * half) * HID + n] = silu(acc[mb][nt][r] + dr[r] * wl + bb);
    }
  }
  __syncthreads();

  v8f acc2[2][8];
#pragma unroll
  for (int mb = 0; mb < 2; ++mb)
#pragma unroll
    for (int nt = 0; nt < 8; ++nt) acc2[mb][nt] = zero8();
  const float* rowS0 = my + m * HID;
  const float* rowS1 = my + (16 + m) * HID;
#pragma unroll
  for (int kb = 0; kb < HID; kb += 32) {
    v16bf a0 = load_a16(rowS0 + kb, half);
    v16bf a1 = load_a16(rowS1 + kb, half);
#pragma unroll
    for (int nt = 0; nt < 8; ++nt) {
      v16bf b = load_b16(W2, HID, nt * 16 + m, kb, half);
      acc2[0][nt] = wmma_bf16(a0, b, acc2[0][nt]);
      acc2[1][nt] = wmma_bf16(a1, b, acc2[1][nt]);
    }
  }

#pragma unroll
  for (int mb = 0; mb < 2; ++mb) {
    int cI[8];
    bool ev[8];
#pragma unroll
    for (int r = 0; r < 8; ++r) {
      int e = e_base + mb * 16 + r + 8 * half;
      ev[r] = (e < E);
      cI[r] = ei[(size_t)E + (ev[r] ? e : 0)];
    }
#pragma unroll
    for (int nt = 0; nt < 8; ++nt) {
      const int n = nt * 16 + m;
      const float bb = b2[n];
#pragma unroll
      for (int r = 0; r < 8; ++r) {
        float val = silu(acc2[mb][nt][r] + bb);
        if (ev[r]) atomicAdd(&maggr[(size_t)cI[r] * HID + n], val);
      }
    }
  }
}

// ------------------------------------------------------------- node MLP (fused)
// h = hn + ( silu([hn, m_aggr*inv_cnt]@nW1^T+nb1) @ nW2^T + nb2 )
__global__ __launch_bounds__(64) void k_node(const float* __restrict__ hn,
                                             const float* __restrict__ maggr,
                                             const float* __restrict__ cnt,
                                             const __bf16* __restrict__ W1,
                                             const float* __restrict__ b1,
                                             const __bf16* __restrict__ W2,
                                             const float* __restrict__ b2,
                                             float* __restrict__ h, int N) {
  __shared__ float smem[2 * 32 * HID];
  const int lane = threadIdx.x & 31, wv = threadIdx.x >> 5;
  const int m = lane & 15, half = lane >> 4;
  const int tiles = N >> 5;
  int tile = blockIdx.x * 2 + wv;
  const bool act = tile < tiles;
  if (!act) tile = tiles - 1;
  const int node0 = tile * 32 + m;
  const int node1 = node0 + 16;
  const float* rowH0 = hn + (size_t)node0 * HID;
  const float* rowH1 = hn + (size_t)node1 * HID;
  const float* rowM0 = maggr + (size_t)node0 * HID;
  const float* rowM1 = maggr + (size_t)node1 * HID;
  const float c0 = cnt[node0], c1 = cnt[node1];
  const float ic0 = c0 > 0.f ? 1.f / fmaxf(c0, 1.f) : 0.f;
  const float ic1 = c1 > 0.f ? 1.f / fmaxf(c1, 1.f) : 0.f;

  v8f acc[2][8];
#pragma unroll
  for (int mb = 0; mb < 2; ++mb)
#pragma unroll
    for (int nt = 0; nt < 8; ++nt) acc[mb][nt] = zero8();

#pragma unroll
  for (int kb = 0; kb < HID; kb += 32) {
    v16bf a0 = load_a16(rowH0 + kb, half);
    v16bf a1 = load_a16(rowH1 + kb, half);
#pragma unroll
    for (int nt = 0; nt < 8; ++nt) {
      v16bf b = load_b16(W1, 256, nt * 16 + m, kb, half);
      acc[0][nt] = wmma_bf16(a0, b, acc[0][nt]);
      acc[1][nt] = wmma_bf16(a1, b, acc[1][nt]);
    }
  }
#pragma unroll
  for (int kb = 0; kb < HID; kb += 32) {
    v16bf a0 = load_a16s(rowM0 + kb, half, ic0);  // scatter-mean scaling on the fly
    v16bf a1 = load_a16s(rowM1 + kb, half, ic1);
#pragma unroll
    for (int nt = 0; nt < 8; ++nt) {
      v16bf b = load_b16(W1, 256, nt * 16 + m, HID + kb, half);
      acc[0][nt] = wmma_bf16(a0, b, acc[0][nt]);
      acc[1][nt] = wmma_bf16(a1, b, acc[1][nt]);
    }
  }

  float* my = smem + wv * 32 * HID;
#pragma unroll
  for (int mb = 0; mb < 2; ++mb)
#pragma unroll
    for (int nt = 0; nt < 8; ++nt) {
      const int n = nt * 16 + m;
      const float bb = b1[n];
#pragma unroll
      for (int r = 0; r < 8; ++r)
        my[(mb * 16 + r + 8 * half) * HID + n] = silu(acc[mb][nt][r] + bb);
    }
  __syncthreads();

  v8f acc2[2][8];
#pragma unroll
  for (int mb = 0; mb < 2; ++mb)
#pragma unroll
    for (int nt = 0; nt < 8; ++nt) acc2[mb][nt] = zero8();
  const float* rowS0 = my + m * HID;
  const float* rowS1 = my + (16 + m) * HID;
#pragma unroll
  for (int kb = 0; kb < HID; kb += 32) {
    v16bf a0 = load_a16(rowS0 + kb, half);
    v16bf a1 = load_a16(rowS1 + kb, half);
#pragma unroll
    for (int nt = 0; nt < 8; ++nt) {
      v16bf b = load_b16(W2, HID, nt * 16 + m, kb, half);
      acc2[0][nt] = wmma_bf16(a0, b, acc2[0][nt]);
      acc2[1][nt] = wmma_bf16(a1, b, acc2[1][nt]);
    }
  }

#pragma unroll
  for (int mb = 0; mb < 2; ++mb)
#pragma unroll
    for (int nt = 0; nt < 8; ++nt) {
      const int n = nt * 16 + m;
      const float bb = b2[n];
#pragma unroll
      for (int r = 0; r < 8; ++r) {
        int nd = tile * 32 + mb * 16 + r + 8 * half;
        if (act) h[(size_t)nd * HID + n] = hn[(size_t)nd * HID + n] + acc2[mb][nt][r] + bb;
      }
    }
}

// ------------------------------------------------------------- output GEMM
// out[N,1024] = h[N,128] @ W_out^T + b_out. wave = (32-row tile, 128-col chunk)
__global__ __launch_bounds__(256) void k_gemm_out(const float* __restrict__ h,
                                                  const __bf16* __restrict__ W,
                                                  const float* __restrict__ bias,
                                                  float* __restrict__ out, int N) {
  const int lane = threadIdx.x & 31, wv = threadIdx.x >> 5;
  const int m = lane & 15, half = lane >> 4;
  const int wave = blockIdx.x * (blockDim.x >> 5) + wv;
  const int nwaves = (N >> 5) * 8;
  if (wave >= nwaves) return;
  const int mt = wave >> 3, chunk = wave & 7;
  const float* rowA0 = h + (size_t)(mt * 32 + m) * HID;
  const float* rowA1 = h + (size_t)(mt * 32 + 16 + m) * HID;
  v8f acc[2][8];
#pragma unroll
  for (int mb = 0; mb < 2; ++mb)
#pragma unroll
    for (int nt = 0; nt < 8; ++nt) acc[mb][nt] = zero8();
#pragma unroll
  for (int kb = 0; kb < HID; kb += 32) {
    v16bf a0 = load_a16(rowA0 + kb, half);
    v16bf a1 = load_a16(rowA1 + kb, half);
#pragma unroll
    for (int nt = 0; nt < 8; ++nt) {
      v16bf b = load_b16(W, HID, chunk * 128 + nt * 16 + m, kb, half);
      acc[0][nt] = wmma_bf16(a0, b, acc[0][nt]);
      acc[1][nt] = wmma_bf16(a1, b, acc[1][nt]);
    }
  }
#pragma unroll
  for (int mb = 0; mb < 2; ++mb)
#pragma unroll
    for (int nt = 0; nt < 8; ++nt) {
      const int n = chunk * 128 + nt * 16 + m;
      const float bi = bias[n];
#pragma unroll
      for (int r = 0; r < 8; ++r) {
        int nd = mt * 32 + mb * 16 + r + 8 * half;
        out[(size_t)nd * CODE + n] = acc[mb][nt][r] + bi;
      }
    }
}

// =========================================================================
extern "C" void kernel_launch(void* const* d_in, const int* in_sizes, int n_in,
                              void* d_out, int out_size, void* d_ws, size_t ws_size,
                              hipStream_t stream) {
  const float* y      = (const float*)d_in[0];
  const float* coords = (const float*)d_in[1];
  const int*   ei     = (const int*)d_in[2];
  const float* W_in   = (const float*)d_in[3];
  const float* b_in   = (const float*)d_in[4];
  const float* gam    = (const float*)d_in[5];
  const float* bet    = (const float*)d_in[6];
  const float* eW1    = (const float*)d_in[7];
  const float* eb1    = (const float*)d_in[8];
  const float* eW2    = (const float*)d_in[9];
  const float* eb2    = (const float*)d_in[10];
  const float* nW1    = (const float*)d_in[11];
  const float* nb1    = (const float*)d_in[12];
  const float* nW2    = (const float*)d_in[13];
  const float* nb2    = (const float*)d_in[14];
  const float* W_out  = (const float*)d_in[15];
  const float* b_out  = (const float*)d_in[16];

  const int N = in_sizes[1] / 3;   // B*n nodes (32768)
  const int E = in_sizes[2] / 2;   // edges

  // ---- workspace carve-out
  char* ws = (char*)d_ws;
  size_t off = 0;
  auto take = [&](size_t bytes) -> char* {
    char* p = ws + off;
    off += (bytes + 255) & ~(size_t)255;
    return p;
  };
  float*  h      = (float*)take((size_t)N * HID * 4);
  float*  hn     = (float*)take((size_t)N * HID * 4);
  float*  maggr  = (float*)take((size_t)N * HID * 4);
  float*  cnt    = (float*)take((size_t)N * 4);
  float*  dist   = (float*)take((size_t)E * 4);
  __bf16* Winbf  = (__bf16*)take((size_t)HID * CODE * 2);
  __bf16* eW1bf  = (__bf16*)take((size_t)NLAYER * HID * 256 * 2);
  __bf16* eW2bf  = (__bf16*)take((size_t)NLAYER * HID * HID * 2);
  __bf16* nW1bf  = (__bf16*)take((size_t)NLAYER * HID * 256 * 2);
  __bf16* nW2bf  = (__bf16*)take((size_t)NLAYER * HID * HID * 2);
  __bf16* Woutbf = (__bf16*)take((size_t)CODE * HID * 2);
  (void)ws_size; (void)n_in; (void)out_size;

  auto cdiv = [](int a, int b) { return (a + b - 1) / b; };

  // ---- prep (re-run every call: deterministic, capture-safe)
  k_convert<<<cdiv(HID * CODE, 256), 256, 0, stream>>>(W_in, Winbf, HID, CODE, CODE);
  k_convert<<<cdiv(NLAYER * HID * 256, 256), 256, 0, stream>>>(eW1, eW1bf, NLAYER * HID, 257, 256);
  k_convert<<<cdiv(NLAYER * HID * HID, 256), 256, 0, stream>>>(eW2, eW2bf, NLAYER * HID, HID, HID);
  k_convert<<<cdiv(NLAYER * HID * 256, 256), 256, 0, stream>>>(nW1, nW1bf, NLAYER * HID, 256, 256);
  k_convert<<<cdiv(NLAYER * HID * HID, 256), 256, 0, stream>>>(nW2, nW2bf, NLAYER * HID, HID, HID);
  k_convert<<<cdiv(CODE * HID, 256), 256, 0, stream>>>(W_out, Woutbf, CODE, HID, HID);

  hipMemsetAsync(cnt, 0, (size_t)N * 4, stream);
  k_edge_geom<<<cdiv(E, 256), 256, 0, stream>>>(ei, coords, dist, cnt, E);

  // ---- input projection
  k_gemm_in<<<cdiv(N / 32, 8), 256, 0, stream>>>(y, Winbf, b_in, h, N);

  // ---- message-passing layers
  for (int l = 0; l < NLAYER; ++l) {
    k_layernorm<<<cdiv(N, 8), 256, 0, stream>>>(h, hn, gam + l * HID, bet + l * HID, N);
    hipMemsetAsync(maggr, 0, (size_t)N * HID * 4, stream);
    const int etiles = cdiv(E, 32);
    k_edge<<<cdiv(etiles, 2), 64, 0, stream>>>(
        hn, ei, dist,
        eW1bf + (size_t)l * HID * 256, eW1 + (size_t)l * HID * 257, eb1 + l * HID,
        eW2bf + (size_t)l * HID * HID, eb2 + l * HID, maggr, E);
    k_node<<<cdiv(N / 32, 2), 64, 0, stream>>>(
        hn, maggr, cnt,
        nW1bf + (size_t)l * HID * 256, nb1 + l * HID,
        nW2bf + (size_t)l * HID * HID, nb2 + l * HID, h, N);
  }

  // ---- output projection
  k_gemm_out<<<cdiv((N / 32) * 8, 8), 256, 0, stream>>>(h, Woutbf, b_out, (float*)d_out, N);
}
